// GlobalMHSAND_47433618817127
// MI455X (gfx1250) — matrix operand order repaired and checked by reading, not compiled, not run on hardware
//
#include <hip/hip_runtime.h>
#include <hip/hip_bf16.h>
#include <math.h>

#define HEADS 8
#define KDIM  32
#define CDIM  256
#define NTOK  1024          // H*W = 32*32
#define BATCH 16
#define ROWS  (BATCH*NTOK)  // 16384 tokens

typedef _Float16 f16;
typedef f16   v16h __attribute__((ext_vector_type(16)));
typedef float v8f  __attribute__((ext_vector_type(8)));

__device__ __forceinline__ v8f wmma16x16x32(v16h a, v16h b, v8f c) {
  // (neg_a, A, neg_b, B, c_mod, C, reuse_a, reuse_b) -> v_wmma_f32_16x16x32_f16
  return __builtin_amdgcn_wmma_f32_16x16x32_f16(false, a, false, b, (short)0, c, false, false);
}

// A-fragment half-slot i (0..15) -> K index within a 32-wide K chunk, lane group g
__device__ __forceinline__ int a_k(int i, int g) { return g*16 + i; }

// ---------------------------------------------------------------------------
// Kernel 0: convert fp32 weights -> f16, transposed ([out_col][k]) so that
// B-fragment K-pairs are contiguous in memory. blockIdx.y selects matrix.
// ---------------------------------------------------------------------------
__global__ void cvt_w_kernel(const float* __restrict__ wq, const float* __restrict__ wk,
                             const float* __restrict__ wv, const float* __restrict__ wp,
                             f16* __restrict__ wqt, f16* __restrict__ wkt,
                             f16* __restrict__ wvt, f16* __restrict__ wpt) {
  int idx = blockIdx.x * 256 + threadIdx.x;       // 0..65535
  const float* src; f16* dst;
  switch (blockIdx.y) {
    case 0:  src = wq; dst = wqt; break;
    case 1:  src = wk; dst = wkt; break;
    case 2:  src = wv; dst = wvt; break;
    default: src = wp; dst = wpt; break;
  }
  int k = idx >> 8, col = idx & 255;              // src is [k][col]
  dst[col * 256 + k] = (f16)src[idx];
}

// ---------------------------------------------------------------------------
// Kernel 1: fused LayerNorm + QKV projection. 16 tokens / block, 128 threads.
// Q,K stored [b,h,n,d]; V stored transposed [b,h,d,n] for contiguous
// B-fragment loads in the attention A@V phase.
// ---------------------------------------------------------------------------
__global__ void __launch_bounds__(128) lnqkv_kernel(
    const float* __restrict__ x, const float* __restrict__ gamma, const float* __restrict__ beta,
    const f16* __restrict__ wqt, const float* __restrict__ bq,
    const f16* __restrict__ wkt, const float* __restrict__ bk,
    const f16* __restrict__ wvt, const float* __restrict__ bv,
    f16* __restrict__ q16, f16* __restrict__ k16, f16* __restrict__ vt16) {
  __shared__ f16 Xs[16][CDIM];                    // 8 KB: LN'd tokens in f16
  int tid = threadIdx.x;
  int r0  = blockIdx.x * 16;

  // ---- LayerNorm: 8 lanes per token (wave32 shuffle reduction) ----
  int tok = tid >> 3, seg = tid & 7;
  const float* xr = x + (size_t)(r0 + tok) * CDIM + seg * 32;
  float vals[32];
  float s = 0.f, s2 = 0.f;
#pragma unroll
  for (int j = 0; j < 32; j++) { float v = xr[j]; vals[j] = v; s += v; s2 += v * v; }
#pragma unroll
  for (int m = 1; m < 8; m <<= 1) { s += __shfl_xor(s, m, 32); s2 += __shfl_xor(s2, m, 32); }
  float mean = s * (1.f / CDIM);
  float var  = s2 * (1.f / CDIM) - mean * mean;
  float rstd = rsqrtf(var + 1e-5f);
#pragma unroll
  for (int j = 0; j < 32; j++) {
    int k = seg * 32 + j;
    Xs[tok][k] = (f16)((vals[j] - mean) * rstd * gamma[k] + beta[k]);
  }
  __syncthreads();

  // ---- QKV GEMM: 48 output tiles (3 matrices x 16 col-tiles) over 4 waves ----
  int wave = tid >> 5, lane = tid & 31;
  int g = lane >> 4, ln = lane & 15;

  // A-fragments are tile-invariant: preload all 8 (K=256) from LDS.
  v16h afr[8];
#pragma unroll
  for (int kb = 0; kb < 8; kb++)
#pragma unroll
    for (int i = 0; i < 16; i++) afr[kb][i] = Xs[ln][kb * 32 + a_k(i, g)];

  for (int tt = wave; tt < 48; tt += 4) {
    int mat = tt >> 4;                            // 0=Q 1=K 2=V
    int ct  = tt & 15;
    const f16*   wt   = (mat == 0) ? wqt : (mat == 1) ? wkt : wvt;
    const float* bias = (mat == 0) ? bq : (mat == 1) ? bk : bv;
    int col = ct * 16 + ln;
    const f16* wcol = wt + col * 256;

    // Preload ALL B-fragments for the K loop -> one big load clause,
    // then run the WMMA chain with descending load waits.
    v16h bfr[8];
#pragma unroll
    for (int kb = 0; kb < 8; kb++)
#pragma unroll
      for (int i = 0; i < 16; i++) bfr[kb][i] = wcol[kb * 32 + g * 16 + i];

    v8f acc = {};
#pragma unroll
    for (int kb = 0; kb < 8; kb++) acc = wmma16x16x32(afr[kb], bfr[kb], acc);

    float bc = bias[col];
    int head = col >> 5, dim = col & 31;
    if (mat < 2) {
      f16* dst = (mat == 0) ? q16 : k16;
#pragma unroll
      for (int v = 0; v < 8; v++) {
        int r = r0 + g * 8 + v;
        int b_ = r >> 10, n = r & 1023;
        dst[(((size_t)(b_ * HEADS + head)) * NTOK + n) * KDIM + dim] = (f16)(acc[v] + bc);
      }
    } else {
      // transposed V: vt16[((b*h+head)*d + dim)*N + n]; 8 consecutive n per lane
#pragma unroll
      for (int v = 0; v < 8; v++) {
        int r = r0 + g * 8 + v;
        int b_ = r >> 10, n = r & 1023;
        vt16[(((size_t)(b_ * HEADS + head)) * KDIM + dim) * NTOK + n] = (f16)(acc[v] + bc);
      }
    }
  }
}

// ---------------------------------------------------------------------------
// Kernel 2: attention per (b, head, 16-query strip).
// LDS: 64KB f32 score strip + 32KB f16 prob strip + 2KB reduction = 98KB.
// ---------------------------------------------------------------------------
__global__ void __launch_bounds__(128) attn_kernel(
    const f16* __restrict__ q16, const f16* __restrict__ k16,
    const f16* __restrict__ vt16, f16* __restrict__ y16) {
  __shared__ float S[16][NTOK];                   // 64 KB raw scores / exps
  __shared__ f16   P[16][NTOK];                   // 32 KB normalized probs (f16)
  __shared__ float Rbuf[2][32][8];                // 2 KB cross-wave partials
  int tid  = threadIdx.x;
  int wave = tid >> 5, lane = tid & 31;
  int g = lane >> 4, ln = lane & 15;

  int bh = blockIdx.x >> 6;                       // 64 query tiles per (b,h)
  int qt = blockIdx.x & 63;
  int b  = bh >> 3, head = bh & 7;
  int qbase = qt * 16;
  const f16* qb  = q16  + (size_t)bh * NTOK * KDIM;
  const f16* kb  = k16  + (size_t)bh * NTOK * KDIM;
  const f16* vtb = vt16 + (size_t)bh * KDIM * NTOK;

  // Loop-invariant Q A-fragment (d == 32 == WMMA K)
  v16h aq;
#pragma unroll
  for (int i = 0; i < 16; i++) aq[i] = qb[(qbase + ln) * KDIM + a_k(i, g)];

  const float scale = 0.17677669529663687f;       // 1/sqrt(32)

  // ---- Phase 1: scores S = scale * Q K^T (16 WMMAs per wave).
  //      Preload all 16 K-fragments (32 x b128) before the WMMA burst. ----
  {
    v16h bfr[16];
#pragma unroll
    for (int j = 0; j < 16; j++) {
      int m0 = (wave + j * 4) * 16;
#pragma unroll
      for (int i = 0; i < 16; i++) bfr[j][i] = kb[(m0 + ln) * KDIM + g * 16 + i];
    }
#pragma unroll
    for (int j = 0; j < 16; j++) {
      int m0 = (wave + j * 4) * 16;
      v8f c = {};
      c = wmma16x16x32(aq, bfr[j], c);
#pragma unroll
      for (int v = 0; v < 8; v++) S[g * 8 + v][m0 + ln] = c[v] * scale;
    }
  }
  __syncthreads();

  // ---- Phase 2: softmax, 8 lanes per row, shuffle-only reductions.
  //      Writes normalized probabilities as f16 into P. ----
  {
    int row = (wave << 2) + (lane >> 3);
    int sg  = lane & 7;
    float* sp = &S[row][sg * 128];
    f16*   pp = &P[row][sg * 128];
    float mx = -3.0e38f;
    for (int i = 0; i < 128; i++) mx = fmaxf(mx, sp[i]);
#pragma unroll
    for (int m = 1; m < 8; m <<= 1) mx = fmaxf(mx, __shfl_xor(mx, m, 32));
    float sum = 0.f;
    for (int i = 0; i < 128; i++) { float e = __expf(sp[i] - mx); sp[i] = e; sum += e; }
#pragma unroll
    for (int m = 1; m < 8; m <<= 1) sum += __shfl_xor(sum, m, 32);
    float inv = 1.f / sum;
    for (int i = 0; i < 128; i++) pp[i] = (f16)(sp[i] * inv);
  }
  __syncthreads();

  // ---- Phase 3: Y = P @ V, all 4 waves: 2 col tiles x 2 K-halves.
  //      Preload all 16 V-fragments (contiguous via transposed V). ----
  {
    int ct = wave & 1;                            // output col tile (dim group)
    int kh = wave >> 1;                           // K half: [kh*512, kh*512+512)
    int dim = ct * 16 + ln;
    const f16* vcol = vtb + (size_t)dim * NTOK + kh * 512;

    v16h bfr[16];
#pragma unroll
    for (int j = 0; j < 16; j++)
#pragma unroll
      for (int i = 0; i < 16; i++) bfr[j][i] = vcol[j * 32 + g * 16 + i];

    v8f acc = {};
#pragma unroll
    for (int j = 0; j < 16; j++) {
      int m0 = kh * 512 + j * 32;
      v16h a;
#pragma unroll
      for (int i = 0; i < 16; i++) a[i] = P[ln][m0 + a_k(i, g)];
      acc = wmma16x16x32(a, bfr[j], acc);
    }

    if (kh == 1) {
#pragma unroll
      for (int v = 0; v < 8; v++) Rbuf[ct][lane][v] = acc[v];
    }
    __syncthreads();
    if (kh == 0) {
#pragma unroll
      for (int v = 0; v < 8; v++) {
        int r = qbase + g * 8 + v;
        float yv = acc[v] + Rbuf[ct][lane][v];
        y16[((size_t)(b * NTOK) + r) * CDIM + head * KDIM + dim] = (f16)yv;
      }
    }
  }
}

// ---------------------------------------------------------------------------
// Kernel 3: output projection Y[16384x256] @ Wp + bp -> fp32 out
// ---------------------------------------------------------------------------
__global__ void __launch_bounds__(128) proj_kernel(
    const f16* __restrict__ y16, const f16* __restrict__ wpt,
    const float* __restrict__ bp, float* __restrict__ out) {
  int tid = threadIdx.x;
  int wave = tid >> 5, lane = tid & 31;
  int g = lane >> 4, ln = lane & 15;
  int r0 = blockIdx.x * 16;
  const f16* yr = y16 + (size_t)r0 * CDIM;

  // A-fragments (token rows) are tile-invariant: preload all 8.
  v16h afr[8];
#pragma unroll
  for (int kb = 0; kb < 8; kb++)
#pragma unroll
    for (int i = 0; i < 16; i++) afr[kb][i] = yr[ln * CDIM + kb * 32 + a_k(i, g)];

  for (int ct = wave; ct < 16; ct += 4) {
    int col = ct * 16 + ln;
    const f16* wcol = wpt + col * 256;

    v16h bfr[8];
#pragma unroll
    for (int kb = 0; kb < 8; kb++)
#pragma unroll
      for (int i = 0; i < 16; i++) bfr[kb][i] = wcol[kb * 32 + g * 16 + i];

    v8f acc = {};
#pragma unroll
    for (int kb = 0; kb < 8; kb++) acc = wmma16x16x32(afr[kb], bfr[kb], acc);

    float bc = bp[col];
#pragma unroll
    for (int v = 0; v < 8; v++)
      out[(size_t)(r0 + g * 8 + v) * CDIM + col] = acc[v] + bc;
  }
}

// ---------------------------------------------------------------------------
extern "C" void kernel_launch(void* const* d_in, const int* in_sizes, int n_in,
                              void* d_out, int out_size, void* d_ws, size_t ws_size,
                              hipStream_t stream) {
  (void)in_sizes; (void)n_in; (void)out_size; (void)ws_size;
  const float* x     = (const float*)d_in[0];
  const float* gamma = (const float*)d_in[1];
  const float* beta  = (const float*)d_in[2];
  const float* Wq = (const float*)d_in[3];
  const float* bq = (const float*)d_in[4];
  const float* Wk = (const float*)d_in[5];
  const float* bk = (const float*)d_in[6];
  const float* Wv = (const float*)d_in[7];
  const float* bv = (const float*)d_in[8];
  const float* Wp = (const float*)d_in[9];
  const float* bp = (const float*)d_in[10];
  float* out = (float*)d_out;

  // workspace carve (halves): 4 x 64K weights, then Q/K/Vt/Y at 4.19M each
  f16* wqt  = (f16*)d_ws;
  f16* wkt  = wqt + 65536;
  f16* wvt  = wkt + 65536;
  f16* wpt  = wvt + 65536;
  f16* q16  = wpt + 65536;
  f16* k16  = q16 + (size_t)ROWS * CDIM;   // 16384*256 = B*h*N*d
  f16* vt16 = k16 + (size_t)ROWS * CDIM;
  f16* y16  = vt16 + (size_t)ROWS * CDIM;

  dim3 gcv(256, 4);
  cvt_w_kernel<<<gcv, 256, 0, stream>>>(Wq, Wk, Wv, Wp, wqt, wkt, wvt, wpt);
  lnqkv_kernel<<<ROWS / 16, 128, 0, stream>>>(x, gamma, beta,
                                              wqt, bq, wkt, bk, wvt, bv,
                                              q16, k16, vt16);
  attn_kernel<<<BATCH * HEADS * (NTOK / 16), 128, 0, stream>>>(q16, k16, vt16, y16);
  proj_kernel<<<ROWS / 16, 128, 0, stream>>>(y16, wpt, bp, out);
}
